// CondMlp_87144886436353
// MI455X (gfx1250) — matrix operand order repaired and checked by reading, not compiled, not run on hardware
//
#include <hip/hip_runtime.h>
#include <hip/hip_bf16.h>

// ---------------------------------------------------------------------------
// CondMLP for MI455X (gfx1250, wave32).
//   Stage 1 (fp32, <2% of FLOPs): XP = x@W_pre+b_pre ; CQ = q@W_emb+b_emb ;
//                                 U = XP@W1x ; V = CQ@W1c + b1
//   Stage 2 (dominant, 34.4 GFLOP): out[b,i,o,:] = gelu(U[b,i,:]+V[b,o,:]) @ W2 + b2
//   Stage 2 uses V_WMMA_F32_16X16X32_BF16, fp32 accumulation.
//   Wave tile 32x32 (4 accumulators) -> 1 fragment load per WMMA.
// ---------------------------------------------------------------------------

typedef __attribute__((ext_vector_type(16))) __bf16 v16bf;
typedef __attribute__((ext_vector_type(8)))  __bf16 v8bf;
typedef __attribute__((ext_vector_type(8)))  float  v8f;

#define Bsz   4
#define NI    256
#define NO    256
#define DH    256

// ---------------------------------------------------------------------------
// Generic fp32 tiled GEMM: C[M,N] = A[M,K] @ B[K,N] (+ bias[N]).
// ---------------------------------------------------------------------------
#define TS 16
__global__ __launch_bounds__(TS * TS)
void gemm_f32(const float* __restrict__ A, const float* __restrict__ B,
              const float* __restrict__ bias, float* __restrict__ C,
              int M, int N, int K) {
  __shared__ float As[TS][TS + 1];
  __shared__ float Bs[TS][TS + 1];
  const int tx = threadIdx.x, ty = threadIdx.y;
  const int col = blockIdx.x * TS + tx;
  const int row = blockIdx.y * TS + ty;
  float acc = 0.0f;
  for (int k0 = 0; k0 < K; k0 += TS) {
    As[ty][tx] = A[row * K + k0 + tx];
    Bs[ty][tx] = B[(k0 + ty) * N + col];
    __syncthreads();
#pragma unroll
    for (int kk = 0; kk < TS; ++kk) acc += As[ty][kk] * Bs[kk][tx];
    __syncthreads();
  }
  if (bias) acc += bias[col];
  C[row * N + col] = acc;
}

// ---------------------------------------------------------------------------
// Convert W2 (fp32, DHxDH row-major) into bf16 laid out exactly as the
// B-fragment register file of V_WMMA_F32_16X16X32_BF16:
//   layout [kstep(8)][ntile(16)][lane(32)][elem(16)]
//   lanes 0-15  : N = lane,      K(elem e) = kstep*32 + e + (e>=8 ? 8 : 0)
//   lanes 16-31 : N = lane - 16, K(elem e) = kstep*32 + 8 + e + (e>=8 ? 8 : 0)
// Each fragment is then one contiguous 32-byte block per lane.
// ---------------------------------------------------------------------------
__global__ __launch_bounds__(256)
void swizzle_w2(const float* __restrict__ W2, __bf16* __restrict__ W2f) {
  const int idx = blockIdx.x * 256 + threadIdx.x;  // 0 .. 65535
  const int k = idx >> 8;                          // row of W2 (K dim)
  const int n = idx & 255;                         // col of W2 (N dim)
  const int kstep = k >> 5, kr = k & 31;
  const int ntile = n >> 4, nl = n & 15;
  const int lane = ((kr >> 3) & 1) ? (16 + nl) : nl;
  const int e    = (kr & 7) + ((kr >> 4) << 3);
  const int dst  = (((kstep * 16 + ntile) * 32 + lane) * 16) + e;
  W2f[dst] = (__bf16)W2[idx];
}

// ---------------------------------------------------------------------------
// Main kernel. Grid (NO/32, NI, B), 256 threads (8 waves).
// Block: fixed (b, i), 32 consecutive o rows, all 256 output columns.
//   Phase 1: h[ol][k] = gelu(U[b,i,k] + V[b,o0+ol,k])  -> bf16 in LDS (32x256)
//   Phase 2: wave w owns a 32(M) x 32(N) tile = 4 accumulators;
//            per k-step: 2 A-frags (LDS) + 2 B-frags (global) -> 4 WMMA.
// ---------------------------------------------------------------------------
__global__ __launch_bounds__(256)
void condmlp_main(const float* __restrict__ U, const float* __restrict__ V,
                  const __bf16* __restrict__ W2f, const float* __restrict__ b2,
                  float* __restrict__ out) {
  const int obig = blockIdx.x;       // 0..7
  const int i    = blockIdx.y;       // 0..255
  const int b    = blockIdx.z;       // 0..3
  const int o0   = obig * 32;
  const int t    = threadIdx.x;      // 0..255

  // Row stride 264 bf16 = 132 dwords; 132 % 64 = 4 -> the 16 rows read by one
  // half-wave hit 16 distinct banks.
  __shared__ __align__(16) __bf16 hs[32][264];

  // ---- Phase 1: build h tile (32 x 256) in bf16 -------------------------
  {
    const float uk  = U[((b * NI + i) * DH) + t];           // column k = t
    const float* vr = V + ((size_t)(b * NO + o0) * DH) + t; // V has b1 folded in
#pragma unroll
    for (int j = 0; j < 32; ++j) {
      const float x = uk + vr[j * DH];
      const float g = 0.5f * x * (1.0f + erff(x * 0.70710678118654752f));
      hs[j][t] = (__bf16)g;
    }
  }
  __syncthreads();

  // ---- Phase 2: WMMA over K = 256 ---------------------------------------
  const int wave = t >> 5;
  const int lane = t & 31;
  const int hi   = lane >> 4;   // 0: lanes 0-15, 1: lanes 16-31
  const int mrow = lane & 15;
  const int nt0  = wave * 2;
  const int nt1  = wave * 2 + 1;

  v8f acc00 = {};   // M rows  0-15, N tile nt0
  v8f acc01 = {};   // M rows  0-15, N tile nt1
  v8f acc10 = {};   // M rows 16-31, N tile nt0
  v8f acc11 = {};   // M rows 16-31, N tile nt1

#pragma unroll
  for (int ks = 0; ks < 8; ++ks) {
    const int k0 = ks * 32;

    // A fragments (16x32 bf16 each): lanes 0-15 row M=lane, elems K {0..7,16..23};
    // lanes 16-31 same rows, elems K {8..15,24..31}.
    v16bf a0, a1;
    {
      const v8bf l0 = *(const v8bf*)(&hs[mrow][k0 + hi * 8]);
      const v8bf h0 = *(const v8bf*)(&hs[mrow][k0 + 16 + hi * 8]);
      const v8bf l1 = *(const v8bf*)(&hs[16 + mrow][k0 + hi * 8]);
      const v8bf h1 = *(const v8bf*)(&hs[16 + mrow][k0 + 16 + hi * 8]);
#pragma unroll
      for (int e = 0; e < 8; ++e) {
        a0[e] = l0[e]; a0[8 + e] = h0[e];
        a1[e] = l1[e]; a1[8 + e] = h1[e];
      }
    }

    // B fragments: one contiguous 32B block per lane from the swizzled W2.
    const v16bf bf0 = *(const v16bf*)(W2f + (((size_t)(ks * 16 + nt0) * 32 + lane) * 16));
    const v16bf bf1 = *(const v16bf*)(W2f + (((size_t)(ks * 16 + nt1) * 32 + lane) * 16));

    acc00 = __builtin_amdgcn_wmma_f32_16x16x32_bf16(false, a0, false, bf0,
                                                    (short)0, acc00, false, false);
    acc01 = __builtin_amdgcn_wmma_f32_16x16x32_bf16(false, a0, false, bf1,
                                                    (short)0, acc01, false, false);
    acc10 = __builtin_amdgcn_wmma_f32_16x16x32_bf16(false, a1, false, bf0,
                                                    (short)0, acc10, false, false);
    acc11 = __builtin_amdgcn_wmma_f32_16x16x32_bf16(false, a1, false, bf1,
                                                    (short)0, acc11, false, false);
  }

  // ---- Epilogue: + b2, store fp32 ---------------------------------------
  // C/D layout: VGPR e -> row M = e + 8*hi (within tile), col N = lane & 15.
  const int ncol0 = nt0 * 16 + mrow;
  const int ncol1 = nt1 * 16 + mrow;
  const float bias0 = b2[ncol0];
  const float bias1 = b2[ncol1];
  float* obase = out + ((((size_t)b * NI + i) * NO + o0) * 256);
#pragma unroll
  for (int e = 0; e < 8; ++e) {
    const int r0 = e + hi * 8;        // rows  0-15 tile
    const int r1 = 16 + e + hi * 8;   // rows 16-31 tile
    obase[(size_t)r0 * 256 + ncol0] = acc00[e] + bias0;
    obase[(size_t)r0 * 256 + ncol1] = acc01[e] + bias1;
    obase[(size_t)r1 * 256 + ncol0] = acc10[e] + bias0;
    obase[(size_t)r1 * 256 + ncol1] = acc11[e] + bias1;
  }
}

// ---------------------------------------------------------------------------
extern "C" void kernel_launch(void* const* d_in, const int* in_sizes, int n_in,
                              void* d_out, int out_size, void* d_ws, size_t ws_size,
                              hipStream_t stream) {
  (void)in_sizes; (void)n_in; (void)out_size; (void)ws_size;

  const float* x     = (const float*)d_in[0];  // (4,256,256)
  const float* query = (const float*)d_in[1];  // (4,256,256)
  const float* W_pre = (const float*)d_in[2];  // (256,256)
  const float* b_pre = (const float*)d_in[3];  // (256,)
  const float* W_emb = (const float*)d_in[4];  // (256,256)
  const float* b_emb = (const float*)d_in[5];  // (256,)
  const float* W1    = (const float*)d_in[6];  // (512,256)
  const float* b1    = (const float*)d_in[7];  // (256,)
  const float* W2    = (const float*)d_in[8];  // (256,256)
  const float* b2    = (const float*)d_in[9];  // (256,)
  float* out = (float*)d_out;

  char* ws = (char*)d_ws;
  float*  XP  = (float*)(ws + (size_t)0 * (1 << 20));   // 1 MB
  float*  CQ  = (float*)(ws + (size_t)1 * (1 << 20));   // 1 MB
  float*  U   = (float*)(ws + (size_t)2 * (1 << 20));   // 1 MB
  float*  V   = (float*)(ws + (size_t)3 * (1 << 20));   // 1 MB
  __bf16* W2f = (__bf16*)(ws + (size_t)4 * (1 << 20));  // 128 KB

  const int M1 = Bsz * NI;   // 1024
  dim3 tb(TS, TS);
  dim3 g1(DH / TS, M1 / TS);

  // Stage 1: fp32 preamble GEMMs (b1 folded into V).
  gemm_f32<<<g1, tb, 0, stream>>>(x,     W_pre,        b_pre,   XP, M1, DH, DH);
  gemm_f32<<<g1, tb, 0, stream>>>(query, W_emb,        b_emb,   CQ, M1, DH, DH);
  gemm_f32<<<g1, tb, 0, stream>>>(XP,    W1,           nullptr, U,  M1, DH, DH);
  gemm_f32<<<g1, tb, 0, stream>>>(CQ,    W1 + DH * DH, b1,      V,  M1, DH, DH);

  // W2 -> bf16 WMMA B-fragment layout.
  swizzle_w2<<<256, 256, 0, stream>>>(W2, W2f);

  // Stage 2: dominant WMMA GEMM (wave tile 32x32, 4 WMMA per k-step).
  dim3 gm(NO / 32, NI, Bsz);
  condmlp_main<<<gm, 256, 0, stream>>>(U, V, W2f, b2, out);
}
